// SwinWindowAttention_50251117363610
// MI455X (gfx1250) — matrix operand order repaired and checked by reading, not compiled, not run on hardware
//
#include <hip/hip_runtime.h>

typedef __attribute__((ext_vector_type(16))) _Float16 v16h;
typedef __attribute__((ext_vector_type(8)))  _Float16 v8h;
typedef __attribute__((ext_vector_type(4)))  _Float16 v4h;
typedef __attribute__((ext_vector_type(8)))  float    v8f;

#define WQKV_HALFS (384*128)
#define WOUT_HALFS (128*128)
#define BIAS_FLOATS (4*49*64)

// LDS strides (padded to dodge bank conflicts; the half strides keep 16-byte alignment)
#define QK_STRIDE 136   // q, k, oA rows (halfs)
#define P_STRIDE  72    // P rows (halfs)
#define VT_STRIDE 72    // vT rows (halfs)
#define S_STRIDE  66    // S rows (floats)

// ---- WMMA fragment loaders (wave32, 16x16x32 f16, per CDNA5 ISA layouts) ----
__device__ __forceinline__ v16h frag_a(const _Float16* base, int stride, int lane) {
  int row = lane & 15;
  int kb  = (lane >> 4) * 8;
  const _Float16* p = base + row * stride + kb;
  v8h lo = *(const v8h*)(p);
  v8h hi = *(const v8h*)(p + 16);
  return __builtin_shufflevector(lo, hi, 0,1,2,3,4,5,6,7,8,9,10,11,12,13,14,15);
}
__device__ __forceinline__ v16h frag_b(const _Float16* baseT, int stride, int lane) {
  int col = lane & 15;
  int kb  = (lane >> 4) * 8;
  const _Float16* p = baseT + col * stride + kb;
  v8h lo = *(const v8h*)(p);
  v8h hi = *(const v8h*)(p + 16);
  return __builtin_shufflevector(lo, hi, 0,1,2,3,4,5,6,7,8,9,10,11,12,13,14,15);
}
__device__ __forceinline__ v8f wmma_f16(v16h a, v16h b, v8f c) {
  return __builtin_amdgcn_wmma_f32_16x16x32_f16(false, a, false, b, (short)0, c, false, false);
}

// ---- Prep kernels ----
__global__ void convert_weights_kernel(const float* __restrict__ qkv_w,
                                       const float* __restrict__ out_w,
                                       _Float16* __restrict__ wq,
                                       _Float16* __restrict__ wo) {
  int i = blockIdx.x * blockDim.x + threadIdx.x;
  if (i < WQKV_HALFS) {
    wq[i] = (_Float16)qkv_w[i];
  } else {
    int j = i - WQKV_HALFS;
    if (j < WOUT_HALFS) wo[j] = (_Float16)out_w[j];
  }
}

__global__ void build_bias_kernel(const float* __restrict__ table,
                                  const int* __restrict__ rel,
                                  float* __restrict__ biasM) {
  int i = blockIdx.x * blockDim.x + threadIdx.x;
  if (i >= BIAS_FLOATS) return;
  int c  = i & 63;
  int hr = i >> 6;
  int r  = hr % 49;
  int h  = hr / 49;
  biasM[i] = (c < 49) ? table[rel[r * 49 + c] * 4 + h] : 0.0f;
}

// ---- Main kernel: one workgroup per window, 8 waves ----
__global__ __launch_bounds__(256) void swin_attn_kernel(
    const float* __restrict__ x, const float* __restrict__ qkv_b,
    const float* __restrict__ out_b, const _Float16* __restrict__ wq,
    const _Float16* __restrict__ wo, const float* __restrict__ biasM,
    float* __restrict__ y)
{
  // region1: xs f16[64][128] -> S f32[4][64][S_STRIDE] -> oA f16[64][QK_STRIDE]
  // region2: q,k f16[64][QK_STRIDE] -> P f16[4][64][P_STRIDE]
  __shared__ __align__(16) unsigned char region1[4 * 64 * S_STRIDE * 4];   // 67584 B
  __shared__ __align__(16) unsigned char region2[4 * 64 * P_STRIDE * 2];   // 36864 B
  __shared__ __align__(16) _Float16 vT[128 * VT_STRIDE];                   // 18432 B

  _Float16* xs = (_Float16*)region1;                    // [64][128]
  float*    S  = (float*)region1;                       // [4][64][S_STRIDE]
  _Float16* oA = (_Float16*)region1;                    // [64][QK_STRIDE]
  _Float16* q  = (_Float16*)region2;                    // [64][QK_STRIDE]
  _Float16* kk = (_Float16*)region2 + 64 * QK_STRIDE;   // [64][QK_STRIDE]
  _Float16* P  = (_Float16*)region2;                    // [4][64][P_STRIDE]

  const int tid  = threadIdx.x;
  const int lane = tid & 31;
  const int wv   = tid >> 5;
  const int hi   = lane >> 4;
  const int cl   = lane & 15;
  const int b    = blockIdx.x;
  const float* xg = x + (size_t)b * 49 * 128;

  // ---- Phase A: x -> f16 LDS (vectorized), zero-pad rows 49..63 ----
  for (int i = tid; i < 64 * 32; i += 256) {       // i indexes float4 within [64][128]
    int r = i >> 5;
    float4 f = make_float4(0.f, 0.f, 0.f, 0.f);
    if (r < 49) f = ((const float4*)xg)[i];
    v4h pk = { (_Float16)f.x, (_Float16)f.y, (_Float16)f.z, (_Float16)f.w };
    *(v4h*)(xs + i * 4) = pk;
  }
  __syncthreads();

  // ---- Phase B: qkv = x @ Wqkv^T + b ; q scaled, v stored transposed ----
  {
    const int mi = wv & 3;
    const int nbase = (wv >> 2) * 12;
    const int nend  = nbase + 12;
    v16h a0 = frag_a(xs + mi * 16 * 128 +  0, 128, lane);
    v16h a1 = frag_a(xs + mi * 16 * 128 + 32, 128, lane);
    v16h a2 = frag_a(xs + mi * 16 * 128 + 64, 128, lane);
    v16h a3 = frag_a(xs + mi * 16 * 128 + 96, 128, lane);

    auto tile = [&](int nt) -> v8f {
      const _Float16* wb = wq + nt * 16 * 128;
      v8f c = {};
      c = wmma_f16(a0, frag_b(wb +  0, 128, lane), c);
      c = wmma_f16(a1, frag_b(wb + 32, 128, lane), c);
      c = wmma_f16(a2, frag_b(wb + 64, 128, lane), c);
      c = wmma_f16(a3, frag_b(wb + 96, 128, lane), c);
      return c;
    };

    const float scale = 0.17677669529663687f; // 1/sqrt(32)
    const int rbase = mi * 16 + 8 * hi;

    // q tiles (nt 0..7)
    for (int nt = nbase; nt < (nend < 8 ? nend : 8); ++nt) {
      v8f c = tile(nt);
      int o = nt * 16 + cl;
      float bias = qkv_b[o];
#pragma unroll
      for (int r = 0; r < 8; ++r)
        q[(rbase + r) * QK_STRIDE + o] = (_Float16)((c[r] + bias) * scale);
    }
    // k tiles (nt 8..15)
    for (int nt = (nbase > 8 ? nbase : 8); nt < (nend < 16 ? nend : 16); ++nt) {
      v8f c = tile(nt);
      int o = nt * 16 + cl;
      float bias = qkv_b[o];
#pragma unroll
      for (int r = 0; r < 8; ++r)
        kk[(rbase + r) * QK_STRIDE + (o - 128)] = (_Float16)(c[r] + bias);
    }
    // v tiles (nt 16..23): element r -> consecutive rows => one packed b128 store
    for (int nt = (nbase > 16 ? nbase : 16); nt < nend; ++nt) {
      v8f c = tile(nt);
      int o = nt * 16 + cl;
      float bias = qkv_b[o];
      v8h pk;
#pragma unroll
      for (int r = 0; r < 8; ++r) pk[r] = (_Float16)(c[r] + bias);
      *(v8h*)(vT + (o - 256) * VT_STRIDE + rbase) = pk;
    }
  }
  __syncthreads();

  // ---- Phase C: S = q_h @ k_h^T (per head, K=32) ----
  {
    const int h  = wv >> 1;
    const int p2 = (wv & 1) * 2;
    for (int mi = 0; mi < 4; ++mi) {
      v16h a = frag_a(q + mi * 16 * QK_STRIDE + h * 32, QK_STRIDE, lane);
#pragma unroll
      for (int j = 0; j < 2; ++j) {
        int ni = p2 + j;
        v16h bb = frag_b(kk + ni * 16 * QK_STRIDE + h * 32, QK_STRIDE, lane);
        v8f c = {};
        c = wmma_f16(a, bb, c);
#pragma unroll
        for (int r = 0; r < 8; ++r) {
          int row = mi * 16 + r + 8 * hi;
          S[(h * 64 + row) * S_STRIDE + ni * 16 + cl] = c[r];
        }
      }
    }
  }
  __syncthreads();

  // ---- Phase D: fp32 softmax + relative-position bias -> P f16 (zero-padded) ----
  {
    if (tid < 196) {
      int h = tid / 49, r = tid % 49;
      float* srow = S + (h * 64 + r) * S_STRIDE;
      const float* brow = biasM + (h * 49 + r) * 64;
      float m = -1e30f;
      for (int c2 = 0; c2 < 49; ++c2) m = fmaxf(m, srow[c2] + brow[c2]);
      float sum = 0.0f;
      for (int c2 = 0; c2 < 49; ++c2) {
        float e = __expf(srow[c2] + brow[c2] - m);
        srow[c2] = e;
        sum += e;
      }
      float inv = 1.0f / sum;
      _Float16* prow = P + (h * 64 + r) * P_STRIDE;
      for (int c2 = 0; c2 < 49; ++c2) prow[c2] = (_Float16)(srow[c2] * inv);
      for (int c2 = 49; c2 < 64; ++c2) prow[c2] = (_Float16)0.0f;
    } else {
      int j = tid - 196;              // 60 threads == 4 heads * 15 pad rows
      int h = j / 15, r = 49 + j % 15;
      _Float16* prow = P + (h * 64 + r) * P_STRIDE;
      for (int c2 = 0; c2 < 64; ++c2) prow[c2] = (_Float16)0.0f;
    }
  }
  __syncthreads();

  // ---- Phase E: out_h = P_h @ v_h (K=64 via two WMMAs) ----
  {
    const int h  = wv >> 1;
    const int mb = (wv & 1) * 2;
    for (int t = 0; t < 2; ++t) {
      int mi = mb + t;
      const _Float16* pb = P + (h * 64 + mi * 16) * P_STRIDE;
      v16h a0 = frag_a(pb +  0, P_STRIDE, lane);
      v16h a1 = frag_a(pb + 32, P_STRIDE, lane);
#pragma unroll
      for (int ni = 0; ni < 2; ++ni) {
        const _Float16* vb = vT + (h * 32 + ni * 16) * VT_STRIDE;
        v8f c = {};
        c = wmma_f16(a0, frag_b(vb +  0, VT_STRIDE, lane), c);
        c = wmma_f16(a1, frag_b(vb + 32, VT_STRIDE, lane), c);
#pragma unroll
        for (int r = 0; r < 8; ++r) {
          int row = mi * 16 + r + 8 * hi;
          oA[row * QK_STRIDE + h * 32 + ni * 16 + cl] = (_Float16)c[r];
        }
      }
    }
  }
  __syncthreads();

  // ---- Phase F: y = out @ Wout^T + b ; store rows < 49 ----
  {
    const int mi = wv & 3;
    const int nbase = (wv >> 2) * 4;
    const _Float16* ob = oA + mi * 16 * QK_STRIDE;
    v16h a0 = frag_a(ob +  0, QK_STRIDE, lane);
    v16h a1 = frag_a(ob + 32, QK_STRIDE, lane);
    v16h a2 = frag_a(ob + 64, QK_STRIDE, lane);
    v16h a3 = frag_a(ob + 96, QK_STRIDE, lane);
    float* yg = y + (size_t)b * 49 * 128;
    for (int nt = nbase; nt < nbase + 4; ++nt) {
      const _Float16* wb = wo + nt * 16 * 128;
      v8f c = {};
      c = wmma_f16(a0, frag_b(wb +  0, 128, lane), c);
      c = wmma_f16(a1, frag_b(wb + 32, 128, lane), c);
      c = wmma_f16(a2, frag_b(wb + 64, 128, lane), c);
      c = wmma_f16(a3, frag_b(wb + 96, 128, lane), c);
      int o = nt * 16 + cl;
      float bias = out_b[o];
      if (mi < 3) {                       // rows 0..47: all valid, no guard
#pragma unroll
        for (int r = 0; r < 8; ++r)
          yg[(mi * 16 + r + 8 * hi) * 128 + o] = c[r] + bias;
      } else {                            // mi==3: only row 48 valid (r==0, hi==0)
        if (hi == 0) yg[48 * 128 + o] = c[0] + bias;
      }
    }
  }
}

extern "C" void kernel_launch(void* const* d_in, const int* in_sizes, int n_in,
                              void* d_out, int out_size, void* d_ws, size_t ws_size,
                              hipStream_t stream) {
  const float* x       = (const float*)d_in[0];
  const float* qkv_w   = (const float*)d_in[1];
  const float* qkv_b   = (const float*)d_in[2];
  const float* out_w   = (const float*)d_in[3];
  const float* out_b   = (const float*)d_in[4];
  const float* btable  = (const float*)d_in[5];
  const int*   rel_idx = (const int*)d_in[6];

  _Float16* wq = (_Float16*)d_ws;
  _Float16* wo = wq + WQKV_HALFS;
  float*    bM = (float*)(wo + WOUT_HALFS);

  convert_weights_kernel<<<(WQKV_HALFS + WOUT_HALFS + 255) / 256, 256, 0, stream>>>(
      qkv_w, out_w, wq, wo);
  build_bias_kernel<<<(BIAS_FLOATS + 255) / 256, 256, 0, stream>>>(btable, rel_idx, bM);

  int nblocks = in_sizes[0] / (49 * 128);   // number of windows (8192)
  swin_attn_kernel<<<nblocks, 256, 0, stream>>>(
      x, qkv_b, out_b, wq, wo, bM, (float*)d_out);
}
